// ChessTransformer_10763188044485
// MI455X (gfx1250) — compile-verified, matrix operand order checked by reading, not run on hardware
//
#include <hip/hip_runtime.h>
#include <hip/hip_bf16.h>
#include <math.h>

typedef __attribute__((ext_vector_type(16))) _Float16 v16h;
typedef __attribute__((ext_vector_type(8)))  _Float16 v8h;
typedef __attribute__((ext_vector_type(4)))  _Float16 v4h;
typedef __attribute__((ext_vector_type(8)))  float    v8f;
typedef __attribute__((ext_vector_type(4)))  float    v4f;

#define D_EMB 1024
#define NHEAD 8
#define DH 128
#define NLAYER 12
#define BATCH 2048
#define SEQ 71
#define MROWS (BATCH * SEQ)   // 145408

#define XS_LD 136   // f16 stride: 272B = 17*16 (aligned), 68 dwords (conflict-free)
#define PS_LD 104   // f16 stride: 208B = 13*16 (aligned), 52 dwords (conflict-free)
#define SS_LD 80    // f32 stride: 80 dwords mod 64 = 16 (conflict-free)
#define AS_LD 72    // f16 stride: 144B = 9*16 (aligned), 36 dwords (conflict-free)

// ---------------- WMMA fragment helpers (wave32, 16x16x32 f16 -> f32) ----------------
__device__ __forceinline__ v16h frag_ld(const _Float16* p) {
  v8h lo = *(const v8h*)p;
  v8h hi = *(const v8h*)(p + 16);
  return __builtin_shufflevector(lo, hi, 0, 1, 2, 3, 4, 5, 6, 7,
                                 8, 9, 10, 11, 12, 13, 14, 15);
}

// A frag from row-major [m][k]; also the B frag when B is staged N-major ([n][k]).
__device__ __forceinline__ v16h frag_a(const _Float16* s, int ld) {
  int lane = threadIdx.x & 31;
  return frag_ld(s + (lane & 15) * ld + ((lane >> 4) * 8));
}

__device__ __forceinline__ v8f wmma_f16(v16h a, v16h b, v8f c) {
  return __builtin_amdgcn_wmma_f32_16x16x32_f16(false, a, false, b, (short)0, c,
                                                false, false);
}

__device__ __forceinline__ float lrelu(float v) { return v > 0.f ? v : 0.2f * v; }

// ---- CDNA5 async global->LDS copy (16B per lane), tracked by ASYNCcnt ----
// LDS address is the byte offset of the destination within the workgroup's LDS
// allocation (kernels below use only dynamic LDS, which starts at offset 0).
__device__ __forceinline__ void async_b128(_Float16* lds_dst, const _Float16* g,
                                           const char* lds_base) {
  unsigned off = (unsigned)((const char*)lds_dst - lds_base);
  asm volatile("global_load_async_to_lds_b128 %0, %1, off"
               :: "v"(off), "v"(g) : "memory");
}
__device__ __forceinline__ void wait_async() {
  asm volatile("s_wait_asynccnt 0x0" ::: "memory");
}

// ---------------- 0) weight prep: convert f32 -> f16, transpose to N-major ----------------
__global__ __launch_bounds__(256)
void transpose128_kernel(const float* __restrict__ in, _Float16* __restrict__ out) {
  size_t mat = blockIdx.x;
  const float* ip = in + mat * (DH * DH);
  _Float16* op = out + mat * (DH * DH);
  for (int i = threadIdx.x; i < DH * DH; i += 256) {
    int d = i >> 7, n = i & 127;
    op[n * DH + d] = (_Float16)ip[i];
  }
}

__global__ __launch_bounds__(256)
void prep_ff1_kernel(const float* __restrict__ in, _Float16* __restrict__ out) {
  size_t mat = blockIdx.x;
  const float* ip = in + mat * (size_t)(D_EMB * DH);
  _Float16* op = out + mat * (size_t)(D_EMB * DH);
  for (int i = threadIdx.x; i < D_EMB * DH; i += 256) {
    int k = i >> 7, n = i & 127;
    op[(size_t)n * D_EMB + k] = (_Float16)ip[i];
  }
}

__global__ __launch_bounds__(256)
void convert_f16_kernel(const float* __restrict__ in, _Float16* __restrict__ out,
                        size_t count) {
  size_t i = (size_t)blockIdx.x * 256 + threadIdx.x;
  if (i < count) out[i] = (_Float16)in[i];
}

// ---------------- 1) embedding + LayerNorm ----------------
__global__ __launch_bounds__(256)
void embed_kernel(const int* __restrict__ fen, const int* __restrict__ move,
                  const float* __restrict__ rank_emb, const float* __restrict__ file_emb,
                  const float* __restrict__ fen_emb, const float* __restrict__ move_emb,
                  const float* __restrict__ g, const float* __restrict__ beta,
                  const float* __restrict__ abs_emb, float* __restrict__ x) {
  int bt = blockIdx.x;
  int b = bt / SEQ, t = bt % SEQ;
  int tid = threadIdx.x;
  __shared__ float red[256];
  float v[4];
  float lsum = 0.f, lsq = 0.f;
#pragma unroll
  for (int i = 0; i < 4; ++i) {
    int d = tid * 4 + i;
    float val;
    if (t < 64) {
      int r = t >> 3, f_ = t & 7;
      float pos = rank_emb[r * D_EMB + d] + file_emb[f_ * D_EMB + d];
      float f0 = fen_emb[fen[b * 133 + t] * D_EMB + d];
      float f1 = fen_emb[fen[b * 133 + 64 + t] * D_EMB + d];
      val = 0.5f * (f0 + f1 + pos);
    } else if (t < 69) {
      val = fen_emb[fen[b * 133 + 128 + (t - 64)] * D_EMB + d];
    } else {
      int m = t - 69;
      int sq = move[b * 2 + m];
      int r = sq >> 3, f_ = sq & 7;
      float pos = rank_emb[r * D_EMB + d] + file_emb[f_ * D_EMB + d];
      val = (pos + move_emb[m * D_EMB + d]) * 0.58f;
    }
    val += abs_emb[t * D_EMB + d];
    v[i] = val;
    lsum += val; lsq += val * val;
  }
  red[tid] = lsum; __syncthreads();
  for (int s = 128; s > 0; s >>= 1) { if (tid < s) red[tid] += red[tid + s]; __syncthreads(); }
  float mean = red[0] * (1.f / D_EMB); __syncthreads();
  red[tid] = lsq; __syncthreads();
  for (int s = 128; s > 0; s >>= 1) { if (tid < s) red[tid] += red[tid + s]; __syncthreads(); }
  float var = red[0] * (1.f / D_EMB) - mean * mean;
  float rstd = rsqrtf(var + 1e-5f);
#pragma unroll
  for (int i = 0; i < 4; ++i) {
    int d = tid * 4 + i;
    x[(size_t)bt * D_EMB + d] = (v[i] - mean) * rstd * g[d] + beta[d];
  }
}

// ---------------- 2) fused per-(batch,head) attention ----------------
// LDS: Qs 21760 | Ks 21760 | Vt 26624 | union{Xs+Wt 56576 / Ss+Ps 42240} = 126720 B
__global__ __launch_bounds__(256)
void attn_kernel(const _Float16* __restrict__ qkvT, float* __restrict__ x, int layer) {
  extern __shared__ char smem[];
  _Float16* Qs = (_Float16*)smem;             // [80][136]
  _Float16* Ks = Qs + 80 * XS_LD;             // [80][136]
  _Float16* Vt = Ks + 80 * XS_LD;             // [128][104]  (V transposed [d][t])
  char* uni = (char*)(Vt + 128 * PS_LD);
  _Float16* Xs = (_Float16*)uni;              // [80][136]
  _Float16* Wt = Xs + 80 * XS_LD;             // [128][136]  (weights N-major [n][k])
  float*    Ss = (float*)uni;                 // [80][80]
  _Float16* Ps = (_Float16*)(uni + 80 * SS_LD * sizeof(float));  // [80][104]

  int b = blockIdx.x, h = blockIdx.y;
  int tid = threadIdx.x;
  int wave = tid >> 5;
  int lane = tid & 31;

  // stage X head-slice: vectorized f32x4 -> packed f16x4
  for (int i = tid; i < 80 * 32; i += 256) {
    int t = i >> 5, c = (i & 31) * 4;
    v4h pk = {};
    if (t < SEQ) {
      v4f xv = *(const v4f*)(x + (size_t)(b * SEQ + t) * D_EMB + h * DH + c);
      pk[0] = (_Float16)xv[0]; pk[1] = (_Float16)xv[1];
      pk[2] = (_Float16)xv[2]; pk[3] = (_Float16)xv[3];
    }
    *(v4h*)(Xs + t * XS_LD + c) = pk;
  }
  // zero V^T pad columns t in [80,96)
  for (int i = tid; i < 128 * 16; i += 256)
    Vt[(i >> 4) * PS_LD + 80 + (i & 15)] = (_Float16)0.f;

  const float qscale = 0.08838834764831845f;  // 1/sqrt(DH)

  // ---- Q, K, V = X * W{q,k,v}; async-copy pre-transposed f16 weights into LDS ----
#pragma unroll
  for (int w = 0; w < 3; ++w) {
    __syncthreads();
    const _Float16* wp = qkvT + (((size_t)layer * 3 + w) * NHEAD + h) * (DH * DH);
    for (int i = tid; i < DH * DH / 8; i += 256) {
      int n = i >> 4, c = (i & 15) * 8;
      async_b128(Wt + n * XS_LD + c, wp + n * DH + c, smem);
    }
    wait_async();
    __syncthreads();
    int nt = wave;
    v16h bf[4];
#pragma unroll
    for (int ks = 0; ks < 4; ++ks)
      bf[ks] = frag_a(Wt + nt * 16 * XS_LD + ks * 32, XS_LD);
    for (int mt = 0; mt < 5; ++mt) {
      v16h af[4];
#pragma unroll
      for (int ks = 0; ks < 4; ++ks)
        af[ks] = frag_a(Xs + mt * 16 * XS_LD + ks * 32, XS_LD);
      v8f acc = {};
#pragma unroll
      for (int ks = 0; ks < 4; ++ks) acc = wmma_f16(af[ks], bf[ks], acc);
      int col = nt * 16 + (lane & 15);          // d index
      int rb = (lane >> 4) * 8;
      if (w == 2) {
        v8h pk;
#pragma unroll
        for (int e = 0; e < 8; ++e) pk[e] = (_Float16)acc[e];
        *(v8h*)(Vt + col * PS_LD + mt * 16 + rb) = pk;
      } else {
        _Float16* dst = (w == 0) ? Qs : Ks;
        float scl = (w == 0) ? qscale : 1.f;
#pragma unroll
        for (int e = 0; e < 8; ++e)
          dst[(mt * 16 + rb + e) * XS_LD + col] = (_Float16)(acc[e] * scl);
      }
    }
  }
  __syncthreads();

  // ---- S = Q * K^T  (B-frag of K^T == A-frag gather of row-major K) ----
  for (int tile = wave; tile < 25; tile += 8) {
    int mt = tile / 5, st = tile % 5;
    v16h af[4], bf[4];
#pragma unroll
    for (int ks = 0; ks < 4; ++ks) {
      af[ks] = frag_a(Qs + mt * 16 * XS_LD + ks * 32, XS_LD);
      bf[ks] = frag_a(Ks + st * 16 * XS_LD + ks * 32, XS_LD);
    }
    v8f acc = {};
#pragma unroll
    for (int ks = 0; ks < 4; ++ks) acc = wmma_f16(af[ks], bf[ks], acc);
    int col = st * 16 + (lane & 15);
    int rb = (lane >> 4) * 8;
#pragma unroll
    for (int e = 0; e < 8; ++e)
      Ss[(mt * 16 + rb + e) * SS_LD + col] = acc[e];
  }
  __syncthreads();

  // ---- softmax rows (one thread per row), f16 P padded to K=96 ----
  if (tid < 80) {
    const float* row = Ss + tid * SS_LD;
    float m = -1e30f;
    for (int s = 0; s < SEQ; ++s) m = fmaxf(m, row[s]);
    float sum = 0.f;
    for (int s = 0; s < SEQ; ++s) sum += __expf(row[s] - m);
    float inv = 1.f / sum;
    _Float16* prow = Ps + tid * PS_LD;
    for (int s = 0; s < SEQ; ++s) prow[s] = (_Float16)(__expf(row[s] - m) * inv);
    for (int s = SEQ; s < 96; ++s) prow[s] = (_Float16)0.f;
  }
  __syncthreads();

  // ---- Y = P * V, residual add into x ----
  {
    int nt = wave;
    v16h bf[3];
#pragma unroll
    for (int ks = 0; ks < 3; ++ks)
      bf[ks] = frag_a(Vt + nt * 16 * PS_LD + ks * 32, PS_LD);
    for (int mt = 0; mt < 5; ++mt) {
      v16h af[3];
#pragma unroll
      for (int ks = 0; ks < 3; ++ks)
        af[ks] = frag_a(Ps + mt * 16 * PS_LD + ks * 32, PS_LD);
      v8f acc = {};
#pragma unroll
      for (int ks = 0; ks < 3; ++ks) acc = wmma_f16(af[ks], bf[ks], acc);
      int col = nt * 16 + (lane & 15);
      int rb = (lane >> 4) * 8;
#pragma unroll
      for (int e = 0; e < 8; ++e) {
        int t = mt * 16 + rb + e;
        if (t < SEQ) {
          size_t idx = ((size_t)(b * SEQ + t)) * D_EMB + h * DH + col;
          x[idx] += acc[e];
        }
      }
    }
  }
}

// ---------------- 3) FF1: y1 = lrelu(x @ ff1[h]) ----------------
// grid (NHEAD, MROWS/64): the 8 head-blocks sharing an A row-tile are adjacent
// in dispatch order -> A panel read from HBM once, 7x from L2.
__global__ __launch_bounds__(256)
void ff1_kernel(const float* __restrict__ x, const _Float16* __restrict__ ff1T,
                _Float16* __restrict__ y1, int layer) {
  extern __shared__ char smem[];
  _Float16* As = (_Float16*)smem;     // [64][72]
  _Float16* Bt = As + 64 * AS_LD;     // [128][72]  ([n][k] chunk)
  int h = blockIdx.x, mb = blockIdx.y;
  int tid = threadIdx.x, wave = tid >> 5, lane = tid & 31;
  size_t m0 = (size_t)mb * 64;
  const _Float16* wT = ff1T + ((size_t)layer * NHEAD + h) * (size_t)(D_EMB * DH);
  v8f acc[4] = {};
  for (int k0 = 0; k0 < D_EMB; k0 += 64) {
    __syncthreads();
    for (int i = tid; i < 128 * 8; i += 256) {   // async weight chunk first
      int n = i >> 3, c = (i & 7) * 8;
      async_b128(Bt + n * AS_LD + c, wT + (size_t)n * D_EMB + k0 + c, smem);
    }
    for (int i = tid; i < 64 * 16; i += 256) {   // A: f32x4 -> f16x4 (needs VALU)
      int r = i >> 4, c = (i & 15) * 4;
      v4f xv = *(const v4f*)(x + (m0 + r) * D_EMB + k0 + c);
      v4h pk;
      pk[0] = (_Float16)xv[0]; pk[1] = (_Float16)xv[1];
      pk[2] = (_Float16)xv[2]; pk[3] = (_Float16)xv[3];
      *(v4h*)(As + r * AS_LD + c) = pk;
    }
    if (k0 + 64 < D_EMB)
      __builtin_prefetch(x + (m0 + (tid & 63)) * D_EMB + k0 + 64 + ((tid >> 6) << 4), 0, 1);
    wait_async();
    __syncthreads();
    v16h bb0 = frag_a(Bt + wave * 16 * AS_LD, AS_LD);
    v16h bb1 = frag_a(Bt + wave * 16 * AS_LD + 32, AS_LD);
#pragma unroll
    for (int mt = 0; mt < 4; ++mt) {
      v16h a0 = frag_a(As + mt * 16 * AS_LD, AS_LD);
      v16h a1 = frag_a(As + mt * 16 * AS_LD + 32, AS_LD);
      acc[mt] = wmma_f16(a0, bb0, acc[mt]);
      acc[mt] = wmma_f16(a1, bb1, acc[mt]);
    }
  }
  int col = wave * 16 + (lane & 15);
  int rb = (lane >> 4) * 8;
  for (int mt = 0; mt < 4; ++mt)
#pragma unroll
    for (int e = 0; e < 8; ++e) {
      size_t m = m0 + mt * 16 + rb + e;
      y1[m * D_EMB + h * DH + col] = (_Float16)lrelu(acc[mt][e]);
    }
}

// ---------------- 4) FF2: x += lrelu(y1_h @ ff2[h]) ----------------
__global__ __launch_bounds__(256)
void ff2_kernel(const _Float16* __restrict__ y1, const _Float16* __restrict__ ff2T,
                float* __restrict__ x, int layer) {
  extern __shared__ char smem[];
  _Float16* As = (_Float16*)smem;     // [64][136]
  _Float16* Wt = As + 64 * XS_LD;     // [128][136]  ([n][k])
  int mb = blockIdx.x, h = blockIdx.y;
  int tid = threadIdx.x, wave = tid >> 5, lane = tid & 31;
  size_t m0 = (size_t)mb * 64;
  const _Float16* wT = ff2T + ((size_t)layer * NHEAD + h) * (DH * DH);
  for (int i = tid; i < 64 * 16; i += 256) {
    int r = i >> 4, c = (i & 15) * 8;
    async_b128(As + r * XS_LD + c, y1 + (m0 + r) * D_EMB + h * DH + c, smem);
  }
  for (int i = tid; i < 128 * 16; i += 256) {
    int n = i >> 4, c = (i & 15) * 8;
    async_b128(Wt + n * XS_LD + c, wT + n * DH + c, smem);
  }
  wait_async();
  __syncthreads();
  v16h bf[4];
#pragma unroll
  for (int ks = 0; ks < 4; ++ks)
    bf[ks] = frag_a(Wt + wave * 16 * XS_LD + ks * 32, XS_LD);
  v8f acc[4] = {};
#pragma unroll
  for (int mt = 0; mt < 4; ++mt) {
    v16h af[4];
#pragma unroll
    for (int ks = 0; ks < 4; ++ks)
      af[ks] = frag_a(As + mt * 16 * XS_LD + ks * 32, XS_LD);
#pragma unroll
    for (int ks = 0; ks < 4; ++ks) acc[mt] = wmma_f16(af[ks], bf[ks], acc[mt]);
  }
  int col = wave * 16 + (lane & 15);
  int rb = (lane >> 4) * 8;
  for (int mt = 0; mt < 4; ++mt)
#pragma unroll
    for (int e = 0; e < 8; ++e) {
      size_t m = m0 + mt * 16 + rb + e;
      x[m * D_EMB + h * DH + col] += lrelu(acc[mt][e]);
    }
}

// ---------------- 5) head LayerNorm over tokens {69,70} concat (f16 out) ----------------
__global__ __launch_bounds__(256)
void headln_kernel(const float* __restrict__ x, const float* __restrict__ g,
                   const float* __restrict__ beta, _Float16* __restrict__ hbuf) {
  int b = blockIdx.x, tid = threadIdx.x;
  __shared__ float red[256];
  float v[8];
  float lsum = 0.f, lsq = 0.f;
#pragma unroll
  for (int i = 0; i < 8; ++i) {
    int j = tid * 8 + i;
    int t = 69 + (j >> 10);
    int d = j & 1023;
    float val = x[((size_t)(b * SEQ + t)) * D_EMB + d];
    v[i] = val; lsum += val; lsq += val * val;
  }
  red[tid] = lsum; __syncthreads();
  for (int s = 128; s > 0; s >>= 1) { if (tid < s) red[tid] += red[tid + s]; __syncthreads(); }
  float mean = red[0] * (1.f / 2048.f); __syncthreads();
  red[tid] = lsq; __syncthreads();
  for (int s = 128; s > 0; s >>= 1) { if (tid < s) red[tid] += red[tid + s]; __syncthreads(); }
  float var = red[0] * (1.f / 2048.f) - mean * mean;
  float rstd = rsqrtf(var + 1e-5f);
#pragma unroll
  for (int i = 0; i < 8; ++i) {
    int j = tid * 8 + i;
    hbuf[(size_t)b * 2048 + j] = (_Float16)((v[i] - mean) * rstd * g[j] + beta[j]);
  }
}

// ---------------- 6) head GEMM: u = lrelu(t @ W1^T + b1) ----------------
// grid (N/128, M/64): N-blocks fastest-varying -> A panels reused out of L2.
__global__ __launch_bounds__(256)
void head_gemm_kernel(const _Float16* __restrict__ hbuf, const _Float16* __restrict__ W1h,
                      const float* __restrict__ b1, float* __restrict__ ubuf) {
  extern __shared__ char smem[];
  _Float16* As = (_Float16*)smem;      // [64][72]
  _Float16* Bs = As + 64 * AS_LD;      // [128][72]  (W1 already [n][k])
  int nb = blockIdx.x, mb = blockIdx.y;
  int tid = threadIdx.x, wave = tid >> 5, lane = tid & 31;
  size_t m0 = (size_t)mb * 64;
  size_t n0 = (size_t)nb * 128;
  v8f acc[4] = {};
  for (int k0 = 0; k0 < 2048; k0 += 64) {
    __syncthreads();
    for (int i = tid; i < 64 * 8; i += 256) {
      int r = i >> 3, c = (i & 7) * 8;
      async_b128(As + r * AS_LD + c, hbuf + (m0 + r) * 2048 + k0 + c, smem);
    }
    for (int i = tid; i < 128 * 8; i += 256) {
      int r = i >> 3, c = (i & 7) * 8;
      async_b128(Bs + r * AS_LD + c, W1h + (n0 + r) * 2048 + k0 + c, smem);
    }
    wait_async();
    __syncthreads();
    v16h bb0 = frag_a(Bs + wave * 16 * AS_LD, AS_LD);
    v16h bb1 = frag_a(Bs + wave * 16 * AS_LD + 32, AS_LD);
#pragma unroll
    for (int mt = 0; mt < 4; ++mt) {
      v16h a0 = frag_a(As + mt * 16 * AS_LD, AS_LD);
      v16h a1 = frag_a(As + mt * 16 * AS_LD + 32, AS_LD);
      acc[mt] = wmma_f16(a0, bb0, acc[mt]);
      acc[mt] = wmma_f16(a1, bb1, acc[mt]);
    }
  }
  size_t n = n0 + wave * 16 + (lane & 15);
  int rb = (lane >> 4) * 8;
  float bias = b1[n];
  for (int mt = 0; mt < 4; ++mt)
#pragma unroll
    for (int e = 0; e < 8; ++e)
      ubuf[(m0 + mt * 16 + rb + e) * 2048 + n] = lrelu(acc[mt][e] + bias);
}

// ---------------- 7) out = sigmoid(u @ W2^T + b2) ----------------
__global__ __launch_bounds__(256)
void head_out_kernel(const float* __restrict__ ubuf, const float* __restrict__ W2,
                     const float* __restrict__ b2, float* __restrict__ out) {
  int b = blockIdx.x, tid = threadIdx.x;
  __shared__ float red[256];
  float s = 0.f;
  for (int j = tid; j < 2048; j += 256) s += ubuf[(size_t)b * 2048 + j] * W2[j];
  red[tid] = s; __syncthreads();
  for (int k = 128; k > 0; k >>= 1) { if (tid < k) red[tid] += red[tid + k]; __syncthreads(); }
  if (tid == 0) {
    float z = red[0] + b2[0];
    out[b] = 1.f / (1.f + __expf(-z));
  }
}

// ---------------- host launcher ----------------
extern "C" void kernel_launch(void* const* d_in, const int* in_sizes, int n_in,
                              void* d_out, int out_size, void* d_ws, size_t ws_size,
                              hipStream_t stream) {
  (void)in_sizes; (void)n_in; (void)out_size; (void)ws_size;
  const int*   fen      = (const int*)d_in[0];
  const int*   move     = (const int*)d_in[1];
  const float* rank_emb = (const float*)d_in[2];
  const float* file_emb = (const float*)d_in[3];
  const float* fen_emb  = (const float*)d_in[4];
  const float* move_emb = (const float*)d_in[5];
  const float* ln_emb_g = (const float*)d_in[6];
  const float* ln_emb_b = (const float*)d_in[7];
  const float* abs_emb  = (const float*)d_in[8];
  const float* qkv      = (const float*)d_in[9];
  const float* ff1      = (const float*)d_in[10];
  const float* ff2      = (const float*)d_in[11];
  const float* ln_out_g = (const float*)d_in[12];
  const float* ln_out_b = (const float*)d_in[13];
  const float* W1       = (const float*)d_in[14];
  const float* b1       = (const float*)d_in[15];
  const float* W2       = (const float*)d_in[16];
  const float* b2       = (const float*)d_in[17];
  float* out = (float*)d_out;

  char* ws = (char*)d_ws;
  size_t off = 0;
  float*    x     = (float*)(ws + off);    off += (size_t)MROWS * D_EMB * 4;
  _Float16* y1    = (_Float16*)(ws + off); off += (size_t)MROWS * D_EMB * 2;
  _Float16* hbuf  = (_Float16*)(ws + off); off += (size_t)BATCH * 2048 * 2;
  float*    ubuf  = (float*)(ws + off);    off += (size_t)BATCH * 2048 * 4;
  _Float16* qkvT  = (_Float16*)(ws + off); off += (size_t)NLAYER * 3 * NHEAD * DH * DH * 2;
  _Float16* ff1T  = (_Float16*)(ws + off); off += (size_t)NLAYER * NHEAD * D_EMB * DH * 2;
  _Float16* ff2T  = (_Float16*)(ws + off); off += (size_t)NLAYER * NHEAD * DH * DH * 2;
  _Float16* W1h   = (_Float16*)(ws + off); off += (size_t)2048 * 2048 * 2;

  // one-time-per-call weight prep (f32 -> f16, N-major)
  transpose128_kernel<<<dim3(NLAYER * 3 * NHEAD), dim3(256), 0, stream>>>(qkv, qkvT);
  transpose128_kernel<<<dim3(NLAYER * NHEAD), dim3(256), 0, stream>>>(ff2, ff2T);
  prep_ff1_kernel<<<dim3(NLAYER * NHEAD), dim3(256), 0, stream>>>(ff1, ff1T);
  convert_f16_kernel<<<dim3((2048 * 2048 + 255) / 256), dim3(256), 0, stream>>>(
      W1, W1h, (size_t)2048 * 2048);

  embed_kernel<<<dim3(MROWS), dim3(256), 0, stream>>>(
      fen, move, rank_emb, file_emb, fen_emb, move_emb, ln_emb_g, ln_emb_b, abs_emb, x);

  const size_t attn_lds = 2 * (80 * XS_LD * 2) + 128 * PS_LD * 2 +
                          (80 * XS_LD * 2 + 128 * XS_LD * 2);  // 126720 B
  const size_t ff1_lds  = (size_t)(64 + 128) * AS_LD * 2;       // 27648 B
  const size_t ff2_lds  = (size_t)(64 + 128) * XS_LD * 2;       // 52224 B
  for (int l = 0; l < NLAYER; ++l) {
    attn_kernel<<<dim3(BATCH, NHEAD), dim3(256), attn_lds, stream>>>(qkvT, x, l);
    ff1_kernel<<<dim3(NHEAD, MROWS / 64), dim3(256), ff1_lds, stream>>>(x, ff1T, y1, l);
    ff2_kernel<<<dim3(MROWS / 64, NHEAD), dim3(256), ff2_lds, stream>>>(y1, ff2T, x, l);
  }

  headln_kernel<<<dim3(BATCH), dim3(256), 0, stream>>>(x, ln_out_g, ln_out_b, hbuf);
  head_gemm_kernel<<<dim3(2048 / 128, BATCH / 64), dim3(256),
                     (size_t)(64 + 128) * AS_LD * 2, stream>>>(hbuf, W1h, b1, ubuf);
  head_out_kernel<<<dim3(BATCH), dim3(256), 0, stream>>>(ubuf, W2, b2, out);
}
